// VanillaSFNN_86474871538278
// MI455X (gfx1250) — matrix operand order repaired and checked by reading, not compiled, not run on hardware
//
#include <hip/hip_runtime.h>

typedef __attribute__((ext_vector_type(16))) _Float16 v16h;
typedef __attribute__((ext_vector_type(8)))  float    v8f;

#define T_STEPS 1000
#define I_DIM   700
#define IP_DIM  704          // zero-padded K for f16 W_h (22 * 32)
#define H_DIM   64
#define O_DIM   20
#define NP_DIM  32           // padded readout rows (20..31 zeroed)
#define M_TILES 63           // ceil(1000/16)
#define K_FULL  672          // 21 full K=32 steps on the A side; remainder guarded

// ---------------- WMMA fragment loaders ------------------------------------
// A 16x32 f16 layout: lanes 0-15: row=lane,   elems 0..15 -> K k..k+15
//                     lanes16-31: row=lane-16, elems 0..15 -> K k+16..k+31
__device__ __forceinline__ v16h loadA16(const float* p, int k, int lane) {
  const int koff = (lane & 16) ? 16 : 0;
  const float4* q  = (const float4*)(p + k + koff);
  const float4* q2 = (const float4*)(p + k + koff + 8);
  float4 a0 = q[0], a1 = q[1], h0 = q2[0], h1 = q2[1];
  v16h r;
  r[0]=(_Float16)a0.x;  r[1]=(_Float16)a0.y;  r[2]=(_Float16)a0.z;  r[3]=(_Float16)a0.w;
  r[4]=(_Float16)a1.x;  r[5]=(_Float16)a1.y;  r[6]=(_Float16)a1.z;  r[7]=(_Float16)a1.w;
  r[8]=(_Float16)h0.x;  r[9]=(_Float16)h0.y;  r[10]=(_Float16)h0.z; r[11]=(_Float16)h0.w;
  r[12]=(_Float16)h1.x; r[13]=(_Float16)h1.y; r[14]=(_Float16)h1.z; r[15]=(_Float16)h1.w;
  return r;
}

__device__ __forceinline__ v16h loadA16g(const float* p, int k, int K, int lane) {
  const int koff = (lane & 16) ? 16 : 0;
  v16h r = {};
#pragma unroll
  for (int e = 0; e < 8; ++e) {
    int k0 = k + koff + e;
    int k1 = k0 + 8;
    r[e]     = (_Float16)((k0 < K) ? p[k0] : 0.f);
    r[e + 8] = (_Float16)((k1 < K) ? p[k1] : 0.f);
  }
  return r;
}

// B 32x16 f16 layout: col = lane&15; lanes 0-15 hold K k..k+15, lanes 16-31 K k+16..k+31.
// Pre-converted f16 weights: one aligned 32B vector load, no cvt, no guards.
__device__ __forceinline__ v16h loadB16h(const _Float16* p, int k, int lane) {
  const int kb = (lane & 16) ? 16 : 0;
  return *(const v16h*)(p + k + kb);
}

// Fallback (no workspace): inline f32 -> f16 conversion
__device__ __forceinline__ v16h loadB16(const float* p, int k, int lane) {
  const int kb = (lane & 16) ? 16 : 0;
  const float4* q = (const float4*)(p + k + kb);
  float4 a0 = q[0], a1 = q[1], a2 = q[2], a3 = q[3];
  v16h r;
  r[0]=(_Float16)a0.x;  r[1]=(_Float16)a0.y;  r[2]=(_Float16)a0.z;  r[3]=(_Float16)a0.w;
  r[4]=(_Float16)a1.x;  r[5]=(_Float16)a1.y;  r[6]=(_Float16)a1.z;  r[7]=(_Float16)a1.w;
  r[8]=(_Float16)a2.x;  r[9]=(_Float16)a2.y;  r[10]=(_Float16)a2.z; r[11]=(_Float16)a2.w;
  r[12]=(_Float16)a3.x; r[13]=(_Float16)a3.y; r[14]=(_Float16)a3.z; r[15]=(_Float16)a3.w;
  return r;
}

__device__ __forceinline__ v16h loadB16g(const float* p, int k, int K, int lane) {
  const int kb = (lane & 16) ? 16 : 0;
  v16h r = {};
#pragma unroll
  for (int e = 0; e < 16; ++e) {
    int kk = k + kb + e;
    r[e] = (_Float16)((kk < K) ? p[kk] : 0.f);
  }
  return r;
}

__device__ __forceinline__ v8f wmma32(v16h a, v16h b, v8f c) {
  return __builtin_amdgcn_wmma_f32_16x16x32_f16(false, a, false, b, (short)0, c,
                                                false, false);
}

// ---------------------------------------------------------------------------
// Pre-kernel: convert weights to zero-padded f16 once.
//   Wh16: [H_DIM][IP_DIM]   (cols 700..703 = 0)
//   Wr16: [NP_DIM][H_DIM]   (rows 20..31  = 0)
// ---------------------------------------------------------------------------
__global__ __launch_bounds__(256) void convert_weights_kernel(
    const float* __restrict__ W_h, const float* __restrict__ W_r,
    _Float16* __restrict__ Wh16, _Float16* __restrict__ Wr16) {
  const int stride = gridDim.x * blockDim.x;
  int idx = blockIdx.x * blockDim.x + threadIdx.x;
  for (int i = idx; i < H_DIM * IP_DIM; i += stride) {
    int r = i / IP_DIM, c = i - r * IP_DIM;
    Wh16[i] = (_Float16)((c < I_DIM) ? W_h[r * I_DIM + c] : 0.f);
  }
  for (int i = idx; i < NP_DIM * H_DIM; i += stride) {
    int r = i >> 6, c = i & 63;
    Wr16[i] = (_Float16)((r < O_DIM) ? W_r[r * H_DIM + c] : 0.f);
  }
}

// ---------------------------------------------------------------------------
// One workgroup per batch element b. currents[1000][64] lives in LDS (250KB).
// Phase 1: WMMA GEMM  currents = x[b] @ W_h^T            (f16 in, f32 acc)
// Phase 2: per-h LIF scan -> spikes in-place in LDS
// Phase 3: WMMA GEMM  ro_in = spikes @ W_r^T + b_r       (in-place cols 0..19)
// Phase 4: wave0 leaky-integrator + masked softmax accumulation
// ---------------------------------------------------------------------------
template <bool PRE>
__global__ __launch_bounds__(256) void snn_fused_kernel(
    const float* __restrict__ x, const float* __restrict__ W_h,
    const float* __restrict__ b_h, const float* __restrict__ W_r,
    const float* __restrict__ b_r, const float* __restrict__ tau_m,
    const _Float16* __restrict__ Wh16, const _Float16* __restrict__ Wr16,
    float* __restrict__ out) {
  extern __shared__ float lds[];  // [T_STEPS][H_DIM]
  const int tid  = threadIdx.x;
  const int lane = tid & 31;
  const int wave = tid >> 5;
  const int col  = lane & 15;
  const int hi8  = (lane & 16) ? 8 : 0;  // C-fragment row offset for lanes 16-31
  const int b    = blockIdx.x;
  const float* xb = x + (size_t)b * T_STEPS * I_DIM;

  // ---- Phase 1: hidden input currents via WMMA ----
  for (int mt = wave; mt < M_TILES; mt += 8) {
    const int t0 = mt << 4;
    int trow = t0 + col;
    if (trow > T_STEPS - 1) trow = T_STEPS - 1;  // clamp tail tile rows
    const float* arow = xb + (size_t)trow * I_DIM;
    v8f c0 = {}, c1 = {}, c2 = {}, c3 = {};
    for (int k = 0; k < K_FULL; k += 32) {
      if (k + 96 < I_DIM) __builtin_prefetch(arow + k + 96, 0, 1);
      v16h a = loadA16(arow, k, lane);
      v16h f0, f1, f2, f3;
      if (PRE) {
        f0 = loadB16h(Wh16 + (col +  0) * IP_DIM, k, lane);
        f1 = loadB16h(Wh16 + (col + 16) * IP_DIM, k, lane);
        f2 = loadB16h(Wh16 + (col + 32) * IP_DIM, k, lane);
        f3 = loadB16h(Wh16 + (col + 48) * IP_DIM, k, lane);
      } else {
        f0 = loadB16(W_h + (size_t)(col +  0) * I_DIM, k, lane);
        f1 = loadB16(W_h + (size_t)(col + 16) * I_DIM, k, lane);
        f2 = loadB16(W_h + (size_t)(col + 32) * I_DIM, k, lane);
        f3 = loadB16(W_h + (size_t)(col + 48) * I_DIM, k, lane);
      }
      c0 = wmma32(a, f0, c0);
      c1 = wmma32(a, f1, c1);
      c2 = wmma32(a, f2, c2);
      c3 = wmma32(a, f3, c3);
    }
    {  // K remainder: 672..699 (A guarded; B supplied by f16 zero padding)
      v16h a = loadA16g(arow, K_FULL, I_DIM, lane);
      v16h f0, f1, f2, f3;
      if (PRE) {
        f0 = loadB16h(Wh16 + (col +  0) * IP_DIM, K_FULL, lane);
        f1 = loadB16h(Wh16 + (col + 16) * IP_DIM, K_FULL, lane);
        f2 = loadB16h(Wh16 + (col + 32) * IP_DIM, K_FULL, lane);
        f3 = loadB16h(Wh16 + (col + 48) * IP_DIM, K_FULL, lane);
      } else {
        f0 = loadB16g(W_h + (size_t)(col +  0) * I_DIM, K_FULL, I_DIM, lane);
        f1 = loadB16g(W_h + (size_t)(col + 16) * I_DIM, K_FULL, I_DIM, lane);
        f2 = loadB16g(W_h + (size_t)(col + 32) * I_DIM, K_FULL, I_DIM, lane);
        f3 = loadB16g(W_h + (size_t)(col + 48) * I_DIM, K_FULL, I_DIM, lane);
      }
      c0 = wmma32(a, f0, c0);
      c1 = wmma32(a, f1, c1);
      c2 = wmma32(a, f2, c2);
      c3 = wmma32(a, f3, c3);
    }
#pragma unroll
    for (int r = 0; r < 8; ++r) {
      int t = t0 + r + hi8;
      if (t < T_STEPS) {
        lds[t * H_DIM + col +  0] = c0[r];
        lds[t * H_DIM + col + 16] = c1[r];
        lds[t * H_DIM + col + 32] = c2[r];
        lds[t * H_DIM + col + 48] = c3[r];
      }
    }
  }
  __syncthreads();

  // ---- Phase 2: LIF scan per hidden neuron (serial in t, parallel in h) ----
  if (tid < H_DIM) {
    const float bh = b_h[tid];
    float v = 0.f;
    for (int t = 0; t < T_STEPS; ++t) {
      float cur = lds[t * H_DIM + tid] + bh;
      v += (cur - v) * 0.5f;              // tau = 2, decay_input
      float s = (v >= 1.0f) ? 1.f : 0.f;  // heaviside(v - v_th)
      v *= (1.f - s);                     // hard reset to 0
      lds[t * H_DIM + tid] = s;           // spikes overwrite currents
    }
  }
  __syncthreads();

  // ---- Phase 3: readout currents ro_in = spikes @ W_r^T + b_r (WMMA) ----
  for (int mt = wave; mt < M_TILES; mt += 8) {
    const int t0 = mt << 4;
    int trow = t0 + col;
    if (trow > T_STEPS - 1) trow = T_STEPS - 1;
    const float* srow = lds + trow * H_DIM;  // LDS reads -> ds_load_b128
    const int n1 = 16 + col;                 // second N tile (cols 16..31; valid < 20)
    v8f d0 = {}, d1 = {};
#pragma unroll
    for (int kk = 0; kk < 2; ++kk) {
      int k = kk << 5;
      v16h a = loadA16(srow, k, lane);
      v16h g0, g1;
      if (PRE) {
        g0 = loadB16h(Wr16 + (col +  0) * H_DIM, k, lane);
        g1 = loadB16h(Wr16 + (col + 16) * H_DIM, k, lane);  // rows >=20 are zero
      } else {
        g0 = loadB16(W_r + (size_t)col * H_DIM, k, lane);
        g1 = loadB16(W_r + (size_t)((n1 < O_DIM) ? n1 : (O_DIM - 1)) * H_DIM, k, lane);
        if (n1 >= O_DIM) g1 = (v16h){};
      }
      d0 = wmma32(a, g0, d0);
      d1 = wmma32(a, g1, d1);
    }
    const float br0 = b_r[col];
    const float br1 = (n1 < O_DIM) ? b_r[n1] : 0.f;
#pragma unroll
    for (int r = 0; r < 8; ++r) {
      int t = t0 + r + hi8;
      if (t < T_STEPS) {
        lds[t * H_DIM + col] = d0[r] + br0;  // in-place over spikes
        if (n1 < O_DIM) lds[t * H_DIM + n1] = d1[r] + br1;
      }
    }
  }
  __syncthreads();

  // ---- Phase 4: leaky readout + masked softmax accumulation (wave 0) ----
  if (wave == 0) {
    const bool act = (lane < O_DIM);
    const int  o   = act ? lane : 0;
    const float beta = 1.f / (1.f + __expf(-tau_m[o]));  // sigmoid
    float vro = 0.f, acc = 0.f;
    for (int t = 0; t < T_STEPS; ++t) {
      float ro = lds[t * H_DIM + o];
      vro = vro * beta + (1.f - beta) * ro;
      float z = act ? vro : -1e30f;
      float m = z;
#pragma unroll
      for (int off = 16; off > 0; off >>= 1) m = fmaxf(m, __shfl_xor(m, off));
      float e = act ? __expf(z - m) : 0.f;
      float s = e;
#pragma unroll
      for (int off = 16; off > 0; off >>= 1) s += __shfl_xor(s, off);
      if (t > 10) acc += e / s;  // mask: t > 10
    }
    if (act) out[(size_t)b * O_DIM + lane] = acc;
  }
}

extern "C" void kernel_launch(void* const* d_in, const int* in_sizes, int n_in,
                              void* d_out, int out_size, void* d_ws, size_t ws_size,
                              hipStream_t stream) {
  (void)in_sizes; (void)n_in; (void)out_size;
  const float* x     = (const float*)d_in[0];
  const float* W_h   = (const float*)d_in[1];
  const float* b_h   = (const float*)d_in[2];
  const float* W_r   = (const float*)d_in[3];
  const float* b_r   = (const float*)d_in[4];
  const float* tau_m = (const float*)d_in[5];
  float* out = (float*)d_out;

  const size_t shmem = (size_t)T_STEPS * H_DIM * sizeof(float);  // 250 KB < 320 KB/WGP
  const size_t need  = ((size_t)H_DIM * IP_DIM + (size_t)NP_DIM * H_DIM) * sizeof(_Float16);

  if (ws_size >= need) {
    _Float16* Wh16 = (_Float16*)d_ws;
    _Float16* Wr16 = Wh16 + (size_t)H_DIM * IP_DIM;
    convert_weights_kernel<<<dim3(32), dim3(256), 0, stream>>>(W_h, W_r, Wh16, Wr16);
    snn_fused_kernel<true><<<dim3(256), dim3(256), shmem, stream>>>(
        x, W_h, b_h, W_r, b_r, tau_m, Wh16, Wr16, out);
  } else {
    snn_fused_kernel<false><<<dim3(256), dim3(256), shmem, stream>>>(
        x, W_h, b_h, W_r, b_r, tau_m, nullptr, nullptr, out);
  }
}
